// GraphConvLayer_1468878815659
// MI455X (gfx1250) — compile-verified, matrix operand order checked by reading, not run on hardware
//
#include <hip/hip_runtime.h>
#include <hip/hip_bf16.h>

typedef __attribute__((ext_vector_type(16))) __bf16 v16bf;
typedef __attribute__((ext_vector_type(8)))  float  v8f;

#define DIM 128
#define FRAG_BYTES 65536   // 8 kt * 8 nt * 32 lanes * 16 bf16 * 2B
#define TILES 4            // 16-edge/node tiles per wave (amortizes LDS staging)

// ---------------------------------------------------------------------------
// Pack a row-major f32 weight matrix W[256][128] into bf16 WMMA B-fragments.
// Fragment storage: frag[((kt*8 + nt)*32 + lane)*16 + e]
//   lane = hi*16 + l ; column n = nt*16 + l
//   e<8 : k = kt*32 + hi*8 + e ; e>=8 : k = kt*32 + 16 + hi*8 + (e-8)
// Mirrors the CDNA5 16-bit 16x32 operand layout (CDNA5 ISA).
// ---------------------------------------------------------------------------
__global__ void pack_w_kernel(const float* __restrict__ W, __bf16* __restrict__ frag) {
    int idx = blockIdx.x * blockDim.x + threadIdx.x;      // 0 .. 32767
    if (idx >= 256 * 128) return;
    int e    = idx & 15;
    int lane = (idx >> 4) & 31;
    int nt   = (idx >> 9) & 7;
    int kt   = idx >> 12;
    int l  = lane & 15;
    int hi = lane >> 4;
    int n = nt * 16 + l;
    int k = kt * 32 + hi * 8 + (e & 7) + ((e >> 3) << 4);
    frag[idx] = (__bf16)W[k * DIM + n];
}

__global__ void zero_kernel(float* __restrict__ p, long long n) {
    long long i = (long long)blockIdx.x * blockDim.x + threadIdx.x;
    if (i < n) p[i] = 0.0f;
}

// ---- raw A tile data (16 f32) + conversion, split so loads pipeline -------
struct RawA { float4 f0, f1, f2, f3; };

__device__ __forceinline__ RawA load_rawA(const float* __restrict__ rp, int k0) {
    RawA r;
    r.f0 = *(const float4*)(rp + k0);
    r.f1 = *(const float4*)(rp + k0 + 4);
    r.f2 = *(const float4*)(rp + k0 + 16);
    r.f3 = *(const float4*)(rp + k0 + 20);
    return r;
}

__device__ __forceinline__ v16bf cvt_a(const RawA& r, float scale) {
    v16bf a;
    a[0]  = (__bf16)(r.f0.x * scale); a[1]  = (__bf16)(r.f0.y * scale);
    a[2]  = (__bf16)(r.f0.z * scale); a[3]  = (__bf16)(r.f0.w * scale);
    a[4]  = (__bf16)(r.f1.x * scale); a[5]  = (__bf16)(r.f1.y * scale);
    a[6]  = (__bf16)(r.f1.z * scale); a[7]  = (__bf16)(r.f1.w * scale);
    a[8]  = (__bf16)(r.f2.x * scale); a[9]  = (__bf16)(r.f2.y * scale);
    a[10] = (__bf16)(r.f2.z * scale); a[11] = (__bf16)(r.f2.w * scale);
    a[12] = (__bf16)(r.f3.x * scale); a[13] = (__bf16)(r.f3.y * scale);
    a[14] = (__bf16)(r.f3.z * scale); a[15] = (__bf16)(r.f3.w * scale);
    return a;
}

__device__ __forceinline__ v16bf load_b(const __bf16* __restrict__ bsm,
                                        int kt, int nt, int lane) {
    return *(const v16bf*)(bsm + (((kt * 8 + nt) * 32) + lane) * 16);
}

// Stage the 64KB packed-weight fragment table into LDS (whole block).
__device__ __forceinline__ const __bf16* stage_weights(const __bf16* __restrict__ wfrag) {
    extern __shared__ char smem_raw[];
    uint4* dst = (uint4*)smem_raw;
    const uint4* src = (const uint4*)wfrag;
#pragma unroll
    for (int i = 0; i < 16; ++i)                  // 256 thr * 16 * 16B = 64KB
        dst[threadIdx.x + 256 * i] = src[threadIdx.x + 256 * i];
    __syncthreads();
    return (const __bf16*)smem_raw;
}

// Core 16x256 @ 256x128 accumulation: A rows from rp0 (k<128) / rp1 (k>=128),
// second half scaled by `scale1`. B fragments from LDS, loaded in groups of 4
// so they occupy distinct registers and overlap with the WMMA bursts.
__device__ __forceinline__ void gemm_tile(
    const __bf16* __restrict__ bsm, int lane, int hi,
    const float* __restrict__ rp0, const float* __restrict__ rp1, float scale1,
    v8f acc[8])
{
    RawA praw = load_rawA(rp0, hi * 8);              // kt = 0
    v16bf a = cvt_a(praw, 1.0f);
#pragma unroll
    for (int kt = 0; kt < 8; ++kt) {
        RawA nraw;
        if (kt < 7) {                                // prefetch next kt's A data
            const float* rpn = ((kt + 1) < 4) ? rp0 : (rp1 - DIM);
            nraw = load_rawA(rpn, (kt + 1) * 32 + hi * 8);
        }
        v16bf bfr0[4], bfr1[4];
#pragma unroll
        for (int j = 0; j < 4; ++j) bfr0[j] = load_b(bsm, kt, j, lane);
#pragma unroll
        for (int j = 0; j < 4; ++j) bfr1[j] = load_b(bsm, kt, 4 + j, lane);
#pragma unroll
        for (int j = 0; j < 4; ++j)
            acc[j] = __builtin_amdgcn_wmma_f32_16x16x32_bf16(
                false, a, false, bfr0[j], (short)0, acc[j], false, false);
        v16bf anext;
        if (kt < 7) anext = cvt_a(nraw, ((kt + 1) < 4) ? 1.0f : scale1);
#pragma unroll
        for (int j = 0; j < 4; ++j)
            acc[4 + j] = __builtin_amdgcn_wmma_f32_16x16x32_bf16(
                false, a, false, bfr1[j], (short)0, acc[4 + j], false, false);
        if (kt < 7) a = anext;
    }
}

// ---------------------------------------------------------------------------
// Fused per-edge message GEMM + relu + scatter-add.
// Wave handles TILES x 16 edges; block = 8 waves. B operands from LDS.
// ---------------------------------------------------------------------------
__global__ __launch_bounds__(256) void msg_kernel(
    const float* __restrict__ x, const long long* __restrict__ eidx,
    const float* __restrict__ eattr, const __bf16* __restrict__ wfrag,
    const float* __restrict__ bmsg, float* __restrict__ summed,
    float* __restrict__ counts, int E)
{
    const __bf16* bsm = stage_weights(wfrag);

    const int lane = threadIdx.x & 31;
    const int wave = threadIdx.x >> 5;
    const int l  = lane & 15;
    const int hi = lane >> 4;

    for (int t = 0; t < TILES; ++t) {
        const long long e0 = ((long long)blockIdx.x * TILES + t) * 128 + wave * 16;
        if (e0 >= E) break;

        long long eA = e0 + l;
        if (eA >= E) eA = E - 1;
        const long long srcn = eidx[eA];             // src node index
        const float* xrow = x + srcn * DIM;
        const float* arow = eattr + eA * DIM;

        v8f acc[8];
#pragma unroll
        for (int i = 0; i < 8; ++i) acc[i] = (v8f){0,0,0,0,0,0,0,0};

        gemm_tile(bsm, lane, hi, xrow, arow, 1.0f, acc);

        // ---- epilogue: bias + relu + scatter-add (lane's rows = hi*8+r) ----
        long long nodes[8];
#pragma unroll
        for (int r = 0; r < 8; ++r) {
            long long e = e0 + hi * 8 + r;
            nodes[r] = (e < E) ? eidx[(long long)E + e] : -1;   // dst node
        }
#pragma unroll
        for (int nt = 0; nt < 8; ++nt) {
            const int n = nt * 16 + l;
            const float bias = bmsg[n];
#pragma unroll
            for (int r = 0; r < 8; ++r) {
                if (nodes[r] >= 0) {
                    float v = acc[nt][r] + bias;
                    v = v > 0.0f ? v : 0.0f;
                    unsafeAtomicAdd(summed + nodes[r] * DIM + n, v);
                }
            }
        }
        if (hi == 0) {
            long long e = e0 + l;
            if (e < E) unsafeAtomicAdd(counts + eidx[(long long)E + e], 1.0f);
        }
    }
}

// ---------------------------------------------------------------------------
// Node update GEMM: h = concat[x, summed/max(count,1)] @ W_upd + b_upd
// Wave handles TILES x 16 node rows. Same pipeline structure.
// ---------------------------------------------------------------------------
__global__ __launch_bounds__(256) void upd_kernel(
    const float* __restrict__ x, const float* __restrict__ summed,
    const float* __restrict__ counts, const __bf16* __restrict__ wfrag,
    const float* __restrict__ bupd, float* __restrict__ h, int N)
{
    const __bf16* bsm = stage_weights(wfrag);

    const int lane = threadIdx.x & 31;
    const int wave = threadIdx.x >> 5;
    const int l  = lane & 15;
    const int hi = lane >> 4;

    for (int t = 0; t < TILES; ++t) {
        const long long n0 = ((long long)blockIdx.x * TILES + t) * 128 + wave * 16;
        if (n0 >= N) break;

        long long node = n0 + l;
        if (node >= N) node = N - 1;
        const float* xrow = x + node * DIM;
        const float* srow = summed + node * DIM;
        const float cnt = counts[node];
        const float inv = 1.0f / fmaxf(cnt, 1.0f);

        v8f acc[8];
#pragma unroll
        for (int i = 0; i < 8; ++i) acc[i] = (v8f){0,0,0,0,0,0,0,0};

        gemm_tile(bsm, lane, hi, xrow, srow, inv, acc);   // mean_msg inline

#pragma unroll
        for (int nt = 0; nt < 8; ++nt) {
            const int n = nt * 16 + l;
            const float bias = bupd[n];
#pragma unroll
            for (int r = 0; r < 8; ++r) {
                long long nd = n0 + hi * 8 + r;
                if (nd < N) h[nd * DIM + n] = acc[nt][r] + bias;
            }
        }
    }
}

// ---------------------------------------------------------------------------
// LayerNorm + residual + relu. One wave (32 lanes x float4) per 128-wide row.
// ---------------------------------------------------------------------------
__global__ __launch_bounds__(256) void ln_kernel(
    const float* __restrict__ h, const float* __restrict__ x,
    const float* __restrict__ gamma, const float* __restrict__ beta,
    float* __restrict__ out, int N)
{
    const int lane = threadIdx.x & 31;
    const int wave = threadIdx.x >> 5;
    const long long row = (long long)blockIdx.x * 8 + wave;
    if (row >= N) return;

    const float4 v = *(const float4*)(h + row * DIM + lane * 4);
    float s  = v.x + v.y + v.z + v.w;
    float s2 = v.x * v.x + v.y * v.y + v.z * v.z + v.w * v.w;
#pragma unroll
    for (int off = 16; off > 0; off >>= 1) {
        s  += __shfl_xor(s,  off, 32);
        s2 += __shfl_xor(s2, off, 32);
    }
    const float mu  = s * (1.0f / 128.0f);
    const float var = s2 * (1.0f / 128.0f) - mu * mu;
    const float rs  = rsqrtf(var + 1e-5f);

    const float4 g  = *(const float4*)(gamma + lane * 4);
    const float4 b  = *(const float4*)(beta + lane * 4);
    const float4 xr = *(const float4*)(x + row * DIM + lane * 4);
    float4 o;
    o.x = fmaxf((v.x - mu) * rs * g.x + b.x + xr.x, 0.0f);
    o.y = fmaxf((v.y - mu) * rs * g.y + b.y + xr.y, 0.0f);
    o.z = fmaxf((v.z - mu) * rs * g.z + b.z + xr.z, 0.0f);
    o.w = fmaxf((v.w - mu) * rs * g.w + b.w + xr.w, 0.0f);
    *(float4*)(out + row * DIM + lane * 4) = o;
}

// ---------------------------------------------------------------------------
extern "C" void kernel_launch(void* const* d_in, const int* in_sizes, int n_in,
                              void* d_out, int out_size, void* d_ws, size_t ws_size,
                              hipStream_t stream) {
    const float*     x     = (const float*)d_in[0];
    const long long* eidx  = (const long long*)d_in[1];
    const float*     eattr = (const float*)d_in[2];
    const float*     Wmsg  = (const float*)d_in[3];
    const float*     bmsg  = (const float*)d_in[4];
    const float*     Wupd  = (const float*)d_in[5];
    const float*     bupd  = (const float*)d_in[6];
    const float*     gamma = (const float*)d_in[7];
    const float*     beta  = (const float*)d_in[8];
    float*           out   = (float*)d_out;

    const int N = in_sizes[0] / DIM;
    const int E = in_sizes[1] / 2;

    char* ws = (char*)d_ws;
    float* summed = (float*)ws;                                  // N*128 f32
    size_t off = (((size_t)N * (DIM + 1) * 4) + 255) & ~(size_t)255;
    float* h = (float*)(ws + off);                               // N*128 f32
    off = ((off + (size_t)N * DIM * 4) + 255) & ~(size_t)255;
    __bf16* wmsgf = (__bf16*)(ws + off);                         // 32768 bf16
    off += FRAG_BYTES;
    __bf16* wupdf = (__bf16*)(ws + off);                         // 32768 bf16
    float* counts = summed + (size_t)N * DIM;                    // N f32 (contiguous)

    // 1) zero segment sums + counts
    {
        long long n = (long long)N * (DIM + 1);
        int blocks = (int)((n + 255) / 256);
        zero_kernel<<<blocks, 256, 0, stream>>>(summed, n);
    }
    // 2) pack both weight matrices into bf16 WMMA fragments
    pack_w_kernel<<<128, 256, 0, stream>>>(Wmsg, wmsgf);
    pack_w_kernel<<<128, 256, 0, stream>>>(Wupd, wupdf);
    // 3) fused edge-message GEMM + scatter (B staged in LDS)
    {
        int perBlock = 128 * TILES;
        int blocks = (E + perBlock - 1) / perBlock;
        msg_kernel<<<blocks, 256, FRAG_BYTES, stream>>>(x, eidx, eattr, wmsgf, bmsg,
                                                        summed, counts, E);
    }
    // 4) node update GEMM (mean computed inline, B staged in LDS)
    {
        int perBlock = 128 * TILES;
        int blocks = (N + perBlock - 1) / perBlock;
        upd_kernel<<<blocks, 256, FRAG_BYTES, stream>>>(x, summed, counts, wupdf,
                                                        bupd, h, N);
    }
    // 5) LayerNorm + residual + relu
    {
        int blocks = (N + 7) / 8;
        ln_kernel<<<blocks, 256, 0, stream>>>(h, x, gamma, beta, out, N);
    }
}
